// SimpleESA3DModel_25924422599451
// MI455X (gfx1250) — matrix operand order repaired and checked by reading, not compiled, not run on hardware
//
#include <hip/hip_runtime.h>
#include <math.h>

#define N_NODES 10000
#define N_EDGES 320000
#define HID 128
#define NODE_DIM 64
#define EDGE_DIM 16
#define NUM_RADIAL 64
#define N_GRAPHS 16
#define NUM_LAYERS 6

typedef __attribute__((ext_vector_type(16))) _Float16 v16h;
typedef __attribute__((ext_vector_type(8)))  _Float16 v8h;
typedef __attribute__((ext_vector_type(8)))  float    v8f;
typedef unsigned int u32x4 __attribute__((ext_vector_type(4)));
typedef int          i32x8 __attribute__((ext_vector_type(8)));
typedef int          i32x4 __attribute__((ext_vector_type(4)));

#if defined(__has_builtin)
#if __has_builtin(__builtin_amdgcn_tensor_load_to_lds) && \
    __has_builtin(__builtin_amdgcn_s_wait_tensorcnt)
#define USE_TDM 1
#endif
#endif
#ifndef USE_TDM
#define USE_TDM 0
#endif

__device__ inline float siluf(float x) {
    // fast native path: v_exp_f32 + v_rcp_f32
    return x * __builtin_amdgcn_rcpf(1.0f + __expf(-x));
}

__device__ inline v8f wmma_f16(v16h a, v16h b, v8f c) {
    return __builtin_amdgcn_wmma_f32_16x16x32_f16(
        false, a, false, b, (short)0, c, false, false);
}

// A fragment (16x32 f16, MxK) from row-major f16 tile with row stride = stride.
__device__ inline v16h load_a_lds(const _Float16* base, int stride, int k0, int lane) {
    int lh = (lane >> 4) & 1;
    int m  = lane & 15;
    const _Float16* p = base + m * stride + k0 + lh * 8;
    v8h lo = *(const v8h*)(p);
    v8h hi = *(const v8h*)(p + 16);
    v16h a;
#pragma unroll
    for (int i = 0; i < 8; ++i) { a[i] = lo[i]; a[8 + i] = hi[i]; }
    return a;
}

// B fragment (32x16 f16, KxN) from W^T stored row-major [N][Kpad] in global.
__device__ inline v16h load_b_glob(const _Float16* wt, int kpad, int n0, int k0, int lane) {
    int lh = (lane >> 4) & 1;
    int n  = lane & 15;
    const _Float16* p = wt + (size_t)(n0 + n) * kpad + k0 + lh * 8;
    v8h lo = *(const v8h*)(p);
    v8h hi = *(const v8h*)(p + 16);
    v16h b;
#pragma unroll
    for (int i = 0; i < 8; ++i) { b[i] = lo[i]; b[8 + i] = hi[i]; }
    return b;
}

__device__ inline void gemm_tile(const _Float16* sIn, int K, const _Float16* wt,
                                 int lane, int n0, v8f& acc) {
    for (int k0 = 0; k0 < K; k0 += 32) {
        if (k0 + 32 < K)
            __builtin_prefetch(wt + (size_t)(n0 + (lane & 15)) * K + k0 + 32, 0, 0);
        v16h a = load_a_lds(sIn, K, k0, lane);
        v16h b = load_b_glob(wt, K, n0, k0, lane);
        acc = wmma_f16(a, b, acc);
    }
}

#if USE_TDM
#if __clang_major__ >= 23
#define TDM_CALL(a, b, c, d)                                         \
    do { i32x8 _z = {0, 0, 0, 0, 0, 0, 0, 0};                        \
         __builtin_amdgcn_tensor_load_to_lds(a, b, c, d, _z, 0); } while (0)
#else
#define TDM_CALL(a, b, c, d) __builtin_amdgcn_tensor_load_to_lds(a, b, c, d, 0)
#endif

// Gather 16 rows of h16 (row length HID f16) into LDS at lds_off with row
// stride 640B (pad 96 DW after each 64-DW row).
__device__ inline void tdm_gather16(unsigned lds_off, const _Float16* gbase,
                                    const int* idx) {
    unsigned long long ga = (unsigned long long)(size_t)(const void*)gbase;
    u32x4 g0;
    g0[0] = 1u | (1u << 31);                       // count=1, gather_mode=1, 16b idx
    g0[1] = lds_off;                               // lds byte address
    g0[2] = (unsigned)ga;                          // global addr lo
    g0[3] = ((unsigned)(ga >> 32) & 0x1FFFFFFu) | (2u << 30);  // hi + type=2
    i32x8 g1;
    g1[0] = (int)((1u << 16) | (1u << 20) | (5u << 22) | (95u << 25));
    g1[1] = (int)((unsigned)HID << 16);            // tensor_dim0 = 128
    g1[2] = (int)((unsigned)N_NODES << 16);        // tensor_dim1 = N_NODES
    g1[3] = (int)((unsigned)HID << 16);            // tile_dim0 = 128
    g1[4] = 16;                                    // tile_dim1 = #indices
    g1[5] = HID;                                   // tensor_dim0_stride = 128
    g1[6] = 0; g1[7] = 0;
    i32x4 g2, g3;
#pragma unroll
    for (int i = 0; i < 4; ++i)
        g2[i] = (idx[2 * i] & 0xFFFF) | (idx[2 * i + 1] << 16);
#pragma unroll
    for (int i = 0; i < 4; ++i)
        g3[i] = (idx[8 + 2 * i] & 0xFFFF) | (idx[8 + 2 * i + 1] << 16);
    TDM_CALL(g0, g1, g2, g3);
}

// Linear copy of nelem f16 from global into LDS, optional LDS padding.
__device__ inline void tdm_linear(unsigned lds_off, const void* gptr, int nelem,
                                  unsigned pad_en, unsigned pad_int, unsigned pad_amt) {
    unsigned long long ga = (unsigned long long)(size_t)gptr;
    u32x4 g0;
    g0[0] = 1u;                                    // count=1, no gather
    g0[1] = lds_off;
    g0[2] = (unsigned)ga;
    g0[3] = ((unsigned)(ga >> 32) & 0x1FFFFFFu) | (2u << 30);
    i32x8 g1;
    g1[0] = (int)((1u << 16) | (pad_en << 20) | (pad_int << 22) | (pad_amt << 25));
    g1[1] = (int)(((unsigned)nelem & 0xFFFFu) << 16);   // tensor_dim0 lo
    g1[2] = (int)((((unsigned)nelem >> 16) & 0xFFFFu) | (1u << 16));  // dim0 hi, dim1=1
    g1[3] = (int)((unsigned)nelem << 16);               // tile_dim0 = nelem
    g1[4] = 0;                                          // tile_dim1/2 unused
    g1[5] = nelem;                                      // stride0
    g1[6] = 0; g1[7] = 0;
    i32x4 gz = {0, 0, 0, 0};
    TDM_CALL(g0, g1, gz, gz);
}
#endif  // USE_TDM

// ---------------- weight conversion: f32 [din,dout] -> f16 [dout][dinpad] ----------------
__global__ void k_convert_wt(const float* __restrict__ w, _Float16* __restrict__ wt,
                             int din, int dout, int dinpad) {
    int idx = blockIdx.x * 256 + threadIdx.x;
    if (idx >= dout * dinpad) return;
    int n = idx / dinpad, k = idx - n * dinpad;
    wt[idx] = (k < din) ? (_Float16)w[(size_t)k * dout + n] : (_Float16)0.0f;
}

__global__ void k_zero(float* __restrict__ p, int n) {
    int idx = blockIdx.x * 256 + threadIdx.x;
    if (idx < n) p[idx] = 0.0f;
}

// ---------------- node embedding ----------------
__global__ __launch_bounds__(256) void k_node_emb(
    const float* __restrict__ x,
    const _Float16* __restrict__ w0t, const float* __restrict__ b0,
    const _Float16* __restrict__ w1t, const float* __restrict__ b1,
    float* __restrict__ h32, _Float16* __restrict__ h16) {
    __shared__ _Float16 sA[16 * NODE_DIM];
    __shared__ _Float16 sH[16 * HID];
    int r0 = blockIdx.x * 16, tid = threadIdx.x;
    for (int idx = tid; idx < 16 * NODE_DIM; idx += 256) {
        int r = idx >> 6, c = idx & 63;
        sA[idx] = (_Float16)x[(size_t)(r0 + r) * NODE_DIM + c];
    }
    __syncthreads();
    int lane = tid & 31, wave = tid >> 5, n0 = wave * 16;
    int lh = (lane >> 4) & 1, nn = lane & 15;
    v8f acc = {};
    gemm_tile(sA, NODE_DIM, w0t, lane, n0, acc);
    float bb = b0[n0 + nn];
#pragma unroll
    for (int r = 0; r < 8; ++r)
        sH[(lh * 8 + r) * HID + n0 + nn] = (_Float16)siluf(acc[r] + bb);
    __syncthreads();
    v8f out = {};
    gemm_tile(sH, HID, w1t, lane, n0, out);
    float b1v = b1[n0 + nn];
#pragma unroll
    for (int r = 0; r < 8; ++r) {
        int row = r0 + lh * 8 + r;
        float v = out[r] + b1v;
        h32[(size_t)row * HID + n0 + nn] = v;
        h16[(size_t)row * HID + n0 + nn] = (_Float16)v;
    }
}

// ---------------- edge embedding ----------------
__global__ __launch_bounds__(256) void k_edge_emb(
    const float* __restrict__ ea,
    const _Float16* __restrict__ w0t, const float* __restrict__ b0,
    const _Float16* __restrict__ w1t, const float* __restrict__ b1,
    _Float16* __restrict__ e16) {
    __shared__ _Float16 sA[16 * 32];
    __shared__ _Float16 sH[16 * HID];
    int e0 = blockIdx.x * 16, tid = threadIdx.x;
    for (int idx = tid; idx < 16 * 32; idx += 256) {
        int r = idx >> 5, c = idx & 31;
        sA[idx] = (c < EDGE_DIM) ? (_Float16)ea[(size_t)(e0 + r) * EDGE_DIM + c]
                                 : (_Float16)0.0f;
    }
    __syncthreads();
    int lane = tid & 31, wave = tid >> 5, n0 = wave * 16;
    int lh = (lane >> 4) & 1, nn = lane & 15;
    v8f acc = {};
    gemm_tile(sA, 32, w0t, lane, n0, acc);
    float bb = b0[n0 + nn];
#pragma unroll
    for (int r = 0; r < 8; ++r)
        sH[(lh * 8 + r) * HID + n0 + nn] = (_Float16)siluf(acc[r] + bb);
    __syncthreads();
    v8f out = {};
    gemm_tile(sH, HID, w1t, lane, n0, out);
    float b1v = b1[n0 + nn];
#pragma unroll
    for (int r = 0; r < 8; ++r)
        e16[(size_t)(e0 + lh * 8 + r) * HID + n0 + nn] = (_Float16)(out[r] + b1v);
}

// ---------------- RBF per edge ----------------
__global__ void k_rbf(const float* __restrict__ pos, const int* __restrict__ ei,
                      const float* __restrict__ centers, const float* __restrict__ widths,
                      _Float16* __restrict__ rbf16) {
    int e = blockIdx.x * 256 + threadIdx.x;
    if (e >= N_EDGES) return;
    int s = ei[e], d = ei[N_EDGES + e];
    float dx = pos[s * 3 + 0] - pos[d * 3 + 0];
    float dy = pos[s * 3 + 1] - pos[d * 3 + 1];
    float dz = pos[s * 3 + 2] - pos[d * 3 + 2];
    float dist = sqrtf(dx * dx + dy * dy + dz * dz);
    float cut = (dist < 10.0f)
                    ? 0.5f * (__cosf(dist * 3.14159265358979323846f / 10.0f) + 1.0f)
                    : 0.0f;
#pragma unroll 4
    for (int k = 0; k < NUM_RADIAL; ++k) {
        float t = dist - centers[k];
        rbf16[(size_t)e * NUM_RADIAL + k] = (_Float16)(cut * __expf(-widths[k] * t * t));
    }
}

// ---------------- per-edge message + edge MLP + scatter-add ----------------
// 64 edges per block (4 M-tiles); B fragments reused across M-tiles.
// LDS: sA 64x320 (A tile; first 64x128 region reused as hidden buffer), sE 64x128.
__global__ __launch_bounds__(256) void k_msg(
    const _Float16* __restrict__ h16, const _Float16* __restrict__ e16,
    const _Float16* __restrict__ rbf16, const int* __restrict__ ei,
    const _Float16* __restrict__ wm0t, const float* __restrict__ bm0,
    const _Float16* __restrict__ wm1t, const float* __restrict__ bm1,
    const _Float16* __restrict__ we0t, const float* __restrict__ be0,
    const _Float16* __restrict__ we1t, const float* __restrict__ be1,
    float* __restrict__ aggr) {
    const int KMSG = 2 * HID + NUM_RADIAL;  // 320
    __shared__ _Float16 sA[64 * (2 * HID + NUM_RADIAL)];
    __shared__ _Float16 sE[64 * HID];
    __shared__ int sSrc[64];
    __shared__ int sDst[64];
    int e0 = blockIdx.x * 64, tid = threadIdx.x;
    if (tid < 64) { sSrc[tid] = ei[e0 + tid]; sDst[tid] = ei[N_EDGES + e0 + tid]; }
    __syncthreads();
#if USE_TDM
    if (tid < 32) {  // wave 0 drives the Tensor Data Mover
        unsigned sA_off = (unsigned)(size_t)(void*)sA;
        unsigned sE_off = (unsigned)(size_t)(void*)sE;
#pragma unroll
        for (int g = 0; g < 4; ++g) {
            unsigned rowb = sA_off + (unsigned)(g * 16 * KMSG * 2);
            tdm_gather16(rowb,                  h16, sDst + g * 16);      // x_i
            tdm_gather16(rowb + 2 * HID,        h16, sSrc + g * 16);      // x_j
        }
        // rbf tile: 64 rows x 64 f16, contiguous in global, 640B LDS row stride
        tdm_linear(sA_off + 2 * 2 * HID, (const char*)rbf16 + (size_t)e0 * NUM_RADIAL * 2,
                   64 * NUM_RADIAL, 1u, 4u, 127u);
        // edge_emb tile: fully contiguous 16 KB
        tdm_linear(sE_off, (const char*)e16 + (size_t)e0 * HID * 2, 64 * HID, 0u, 0u, 0u);
        __builtin_amdgcn_s_wait_tensorcnt(0);
    }
#else
    for (int idx = tid; idx < 64 * KMSG; idx += 256) {
        int r = idx / KMSG, c = idx - r * KMSG;
        _Float16 v;
        if (c < HID)          v = h16[(size_t)sDst[r] * HID + c];
        else if (c < 2 * HID) v = h16[(size_t)sSrc[r] * HID + (c - HID)];
        else                  v = rbf16[(size_t)(e0 + r) * NUM_RADIAL + (c - 2 * HID)];
        sA[idx] = v;
    }
    for (int idx = tid; idx < 64 * HID; idx += 256) {
        int r = idx >> 7, c = idx & 127;
        sE[idx] = e16[(size_t)(e0 + r) * HID + c];
    }
#endif
    __syncthreads();
    int lane = tid & 31, wave = tid >> 5, n0 = wave * 16;
    int lh = (lane >> 4) & 1, nn = lane & 15;
    _Float16* sAh = sA;  // 64x128 hidden buffer aliasing start of sA

    // ---- msg MLP layer 1: one B load feeds 4 WMMAs ----
    v8f accM[4] = {};
    for (int k0 = 0; k0 < KMSG; k0 += 32) {
        v16h b = load_b_glob(wm0t, KMSG, n0, k0, lane);
#pragma unroll
        for (int m = 0; m < 4; ++m) {
            v16h a = load_a_lds(sA + m * 16 * KMSG, KMSG, k0, lane);
            accM[m] = wmma_f16(a, b, accM[m]);
        }
    }
    __syncthreads();  // everyone done reading sA before hidden overwrite
    float bb = bm0[n0 + nn];
#pragma unroll
    for (int m = 0; m < 4; ++m)
#pragma unroll
        for (int r = 0; r < 8; ++r)
            sAh[(m * 16 + lh * 8 + r) * HID + n0 + nn] = (_Float16)siluf(accM[m][r] + bb);
    __syncthreads();

    // ---- msg MLP layer 2 ----
    v8f out[4] = {};
    for (int k0 = 0; k0 < HID; k0 += 32) {
        v16h b = load_b_glob(wm1t, HID, n0, k0, lane);
#pragma unroll
        for (int m = 0; m < 4; ++m) {
            v16h a = load_a_lds(sAh + m * 16 * HID, HID, k0, lane);
            out[m] = wmma_f16(a, b, out[m]);
        }
    }
    __syncthreads();  // done reading hidden before edge-path overwrite

    // ---- edge MLP layer 1 ----
    v8f accE[4] = {};
    for (int k0 = 0; k0 < HID; k0 += 32) {
        v16h b = load_b_glob(we0t, HID, n0, k0, lane);
#pragma unroll
        for (int m = 0; m < 4; ++m) {
            v16h a = load_a_lds(sE + m * 16 * HID, HID, k0, lane);
            accE[m] = wmma_f16(a, b, accE[m]);
        }
    }
    float bbe = be0[n0 + nn];
#pragma unroll
    for (int m = 0; m < 4; ++m)
#pragma unroll
        for (int r = 0; r < 8; ++r)
            sAh[(m * 16 + lh * 8 + r) * HID + n0 + nn] = (_Float16)siluf(accE[m][r] + bbe);
    __syncthreads();

    // ---- edge MLP layer 2, accumulated on top of msg result ----
    for (int k0 = 0; k0 < HID; k0 += 32) {
        v16h b = load_b_glob(we1t, HID, n0, k0, lane);
#pragma unroll
        for (int m = 0; m < 4; ++m) {
            v16h a = load_a_lds(sAh + m * 16 * HID, HID, k0, lane);
            out[m] = wmma_f16(a, b, out[m]);
        }
    }
    float bsum = bm1[n0 + nn] + be1[n0 + nn];
#pragma unroll
    for (int m = 0; m < 4; ++m)
#pragma unroll
        for (int r = 0; r < 8; ++r) {
            int row = m * 16 + lh * 8 + r;
            atomicAdd(&aggr[(size_t)sDst[row] * HID + n0 + nn], out[m][r] + bsum);
        }
}

// ---------------- node update: h += MLP2(upd, concat(aggr, h)) ----------------
__global__ __launch_bounds__(256) void k_upd(
    const float* __restrict__ aggr, float* __restrict__ h32, _Float16* __restrict__ h16,
    const _Float16* __restrict__ w0t, const float* __restrict__ b0,
    const _Float16* __restrict__ w1t, const float* __restrict__ b1) {
    const int KUPD = 2 * HID;  // 256
    __shared__ _Float16 sA[16 * 2 * HID];
    __shared__ _Float16 sH[16 * HID];
    int r0 = blockIdx.x * 16, tid = threadIdx.x;
    for (int idx = tid; idx < 16 * KUPD; idx += 256) {
        int r = idx >> 8, c = idx & 255;
        sA[idx] = (c < HID) ? (_Float16)aggr[(size_t)(r0 + r) * HID + c]
                            : h16[(size_t)(r0 + r) * HID + (c - HID)];
    }
    __syncthreads();
    int lane = tid & 31, wave = tid >> 5, n0 = wave * 16;
    int lh = (lane >> 4) & 1, nn = lane & 15;
    v8f acc = {};
    gemm_tile(sA, KUPD, w0t, lane, n0, acc);
    float bb = b0[n0 + nn];
#pragma unroll
    for (int r = 0; r < 8; ++r)
        sH[(lh * 8 + r) * HID + n0 + nn] = (_Float16)siluf(acc[r] + bb);
    __syncthreads();
    v8f out = {};
    gemm_tile(sH, HID, w1t, lane, n0, out);
    float b1v = b1[n0 + nn];
#pragma unroll
    for (int r = 0; r < 8; ++r) {
        int row = r0 + lh * 8 + r;
        float v = h32[(size_t)row * HID + n0 + nn] + out[r] + b1v;
        h32[(size_t)row * HID + n0 + nn] = v;
        h16[(size_t)row * HID + n0 + nn] = (_Float16)v;
    }
}

// ---------------- graph pooling + output head ----------------
__global__ void k_pool(const float* __restrict__ h32, const int* __restrict__ batch,
                       float* __restrict__ gsum, float* __restrict__ cnt) {
    int idx = blockIdx.x * 256 + threadIdx.x;
    if (idx >= N_NODES * HID) return;
    int node = idx >> 7, c = idx & 127;
    int g = batch[node];
    atomicAdd(&gsum[g * HID + c], h32[idx]);
    if (c == 0) atomicAdd(&cnt[g], 1.0f);
}

__global__ __launch_bounds__(256) void k_out(
    const float* __restrict__ gsum, const float* __restrict__ cnt,
    const float* __restrict__ w0, const float* __restrict__ b0,
    const float* __restrict__ w1, const float* __restrict__ b1,
    float* __restrict__ out) {
    __shared__ float sG[N_GRAPHS * HID];
    __shared__ float sHid[N_GRAPHS * HID];
    int tid = threadIdx.x;
    for (int idx = tid; idx < N_GRAPHS * HID; idx += 256) {
        int g = idx >> 7;
        sG[idx] = gsum[idx] / fmaxf(cnt[g], 1.0f);
    }
    __syncthreads();
    for (int idx = tid; idx < N_GRAPHS * HID; idx += 256) {
        int g = idx >> 7, c = idx & 127;
        float acc = b0[c];
        for (int k = 0; k < HID; ++k) acc += sG[g * HID + k] * w0[k * HID + c];
        sHid[idx] = siluf(acc);
    }
    __syncthreads();
    if (tid < N_GRAPHS) {
        float acc = b1[0];
        for (int k = 0; k < HID; ++k) acc += sHid[tid * HID + k] * w1[k];
        out[tid] = acc;
    }
}

extern "C" void kernel_launch(void* const* d_in, const int* in_sizes, int n_in,
                              void* d_out, int out_size, void* d_ws, size_t ws_size,
                              hipStream_t stream) {
    (void)in_sizes; (void)n_in; (void)out_size; (void)ws_size;
    const float* x         = (const float*)d_in[0];
    const float* pos       = (const float*)d_in[1];
    const int*   ei        = (const int*)d_in[2];
    const int*   batch     = (const int*)d_in[4];
    const float* edge_attr = (const float*)d_in[5];

    const int P = 6;
    const float* ee_b0 = (const float*)d_in[P + 0];
    const float* ee_w0 = (const float*)d_in[P + 1];
    const float* ee_b1 = (const float*)d_in[P + 2];
    const float* ee_w1 = (const float*)d_in[P + 3];
    const float *le_b0[NUM_LAYERS], *le_w0[NUM_LAYERS], *le_b1[NUM_LAYERS], *le_w1[NUM_LAYERS];
    const float *lm_b0[NUM_LAYERS], *lm_w0[NUM_LAYERS], *lm_b1[NUM_LAYERS], *lm_w1[NUM_LAYERS];
    const float *lu_b0[NUM_LAYERS], *lu_w0[NUM_LAYERS], *lu_b1[NUM_LAYERS], *lu_w1[NUM_LAYERS];
    for (int l = 0; l < NUM_LAYERS; ++l) {
        int B = P + 4 + 12 * l;
        le_b0[l] = (const float*)d_in[B + 0];  le_w0[l] = (const float*)d_in[B + 1];
        le_b1[l] = (const float*)d_in[B + 2];  le_w1[l] = (const float*)d_in[B + 3];
        lm_b0[l] = (const float*)d_in[B + 4];  lm_w0[l] = (const float*)d_in[B + 5];
        lm_b1[l] = (const float*)d_in[B + 6];  lm_w1[l] = (const float*)d_in[B + 7];
        lu_b0[l] = (const float*)d_in[B + 8];  lu_w0[l] = (const float*)d_in[B + 9];
        lu_b1[l] = (const float*)d_in[B + 10]; lu_w1[l] = (const float*)d_in[B + 11];
    }
    const float* ne_b0 = (const float*)d_in[P + 76];
    const float* ne_w0 = (const float*)d_in[P + 77];
    const float* ne_b1 = (const float*)d_in[P + 78];
    const float* ne_w1 = (const float*)d_in[P + 79];
    const float* o_b0  = (const float*)d_in[P + 80];
    const float* o_w0  = (const float*)d_in[P + 81];
    const float* o_b1  = (const float*)d_in[P + 82];
    const float* o_w1  = (const float*)d_in[P + 83];
    const float* rbf_c = (const float*)d_in[P + 84];
    const float* rbf_w = (const float*)d_in[P + 85];

    size_t off = 0;
    auto carve = [&](size_t bytes) -> void* {
        void* p = (char*)d_ws + off;
        off = (off + bytes + 255) & ~(size_t)255;
        return p;
    };
    float*     h32   = (float*)carve((size_t)N_NODES * HID * 4);
    _Float16*  h16   = (_Float16*)carve((size_t)N_NODES * HID * 2);
    float*     aggr  = (float*)carve((size_t)N_NODES * HID * 4);
    _Float16*  e16   = (_Float16*)carve((size_t)N_EDGES * HID * 2);
    _Float16*  rbf16 = (_Float16*)carve((size_t)N_EDGES * NUM_RADIAL * 2);
    float*     gbuf  = (float*)carve((N_GRAPHS * HID + N_GRAPHS) * 4);
    float*     gsum  = gbuf;
    float*     cnt   = gbuf + N_GRAPHS * HID;
    _Float16*  wn0t  = (_Float16*)carve((size_t)HID * NODE_DIM * 2);
    _Float16*  wn1t  = (_Float16*)carve((size_t)HID * HID * 2);
    _Float16*  wee0t = (_Float16*)carve((size_t)HID * 32 * 2);
    _Float16*  wee1t = (_Float16*)carve((size_t)HID * HID * 2);
    _Float16 *we0t[NUM_LAYERS], *we1t[NUM_LAYERS], *wm0t[NUM_LAYERS], *wm1t[NUM_LAYERS],
             *wu0t[NUM_LAYERS], *wu1t[NUM_LAYERS];
    for (int l = 0; l < NUM_LAYERS; ++l) {
        we0t[l] = (_Float16*)carve((size_t)HID * HID * 2);
        we1t[l] = (_Float16*)carve((size_t)HID * HID * 2);
        wm0t[l] = (_Float16*)carve((size_t)HID * 320 * 2);
        wm1t[l] = (_Float16*)carve((size_t)HID * HID * 2);
        wu0t[l] = (_Float16*)carve((size_t)HID * 256 * 2);
        wu1t[l] = (_Float16*)carve((size_t)HID * HID * 2);
    }

    auto conv = [&](const float* w, _Float16* wt, int din, int dout, int dinpad) {
        int n = dout * dinpad;
        k_convert_wt<<<(n + 255) / 256, 256, 0, stream>>>(w, wt, din, dout, dinpad);
    };
    conv(ne_w0, wn0t, NODE_DIM, HID, NODE_DIM);
    conv(ne_w1, wn1t, HID, HID, HID);
    conv(ee_w0, wee0t, EDGE_DIM, HID, 32);
    conv(ee_w1, wee1t, HID, HID, HID);
    for (int l = 0; l < NUM_LAYERS; ++l) {
        conv(le_w0[l], we0t[l], HID, HID, HID);
        conv(le_w1[l], we1t[l], HID, HID, HID);
        conv(lm_w0[l], wm0t[l], 320, HID, 320);
        conv(lm_w1[l], wm1t[l], HID, HID, HID);
        conv(lu_w0[l], wu0t[l], 256, HID, 256);
        conv(lu_w1[l], wu1t[l], HID, HID, HID);
    }

    k_node_emb<<<N_NODES / 16, 256, 0, stream>>>(x, wn0t, ne_b0, wn1t, ne_b1, h32, h16);
    k_edge_emb<<<N_EDGES / 16, 256, 0, stream>>>(edge_attr, wee0t, ee_b0, wee1t, ee_b1, e16);
    k_rbf<<<(N_EDGES + 255) / 256, 256, 0, stream>>>(pos, ei, rbf_c, rbf_w, rbf16);

    for (int l = 0; l < NUM_LAYERS; ++l) {
        int na = N_NODES * HID;
        k_zero<<<(na + 255) / 256, 256, 0, stream>>>(aggr, na);
        k_msg<<<N_EDGES / 64, 256, 0, stream>>>(h16, e16, rbf16, ei,
                                                wm0t[l], lm_b0[l], wm1t[l], lm_b1[l],
                                                we0t[l], le_b0[l], we1t[l], le_b1[l], aggr);
        k_upd<<<N_NODES / 16, 256, 0, stream>>>(aggr, h32, h16,
                                                wu0t[l], lu_b0[l], wu1t[l], lu_b1[l]);
    }

    int ng = N_GRAPHS * HID + N_GRAPHS;
    k_zero<<<(ng + 255) / 256, 256, 0, stream>>>(gbuf, ng);
    k_pool<<<(N_NODES * HID + 255) / 256, 256, 0, stream>>>(h32, batch, gsum, cnt);
    k_out<<<1, 256, 0, stream>>>(gsum, cnt, o_w0, o_b0, o_w1, o_b1, (float*)d_out);
}
